// FullModelV1_28819230556831
// MI455X (gfx1250) — matrix operand order repaired and checked by reading, not compiled, not run on hardware
//
#include <hip/hip_runtime.h>

// ---------------------------------------------------------------------------
// Types for CDNA5 WMMA (gfx1250, wave32)
// ---------------------------------------------------------------------------
typedef __attribute__((ext_vector_type(16))) __bf16          bf16x16;
typedef __attribute__((ext_vector_type(8)))  float           v8f;
typedef __attribute__((ext_vector_type(8)))  unsigned short  ushort8;
typedef __attribute__((ext_vector_type(16))) unsigned short  ushort16;

union BF16Frag { ushort16 u; bf16x16 b; };

#define SHUF16(a, b) __builtin_shufflevector((a), (b), 0,1,2,3,4,5,6,7,8,9,10,11,12,13,14,15)

static constexpr int B_   = 256;
static constexpr int T_   = 75;
static constexpr int HID_ = 1024;
static constexpr int G3_  = 3072;
static constexpr int NUM_DAYS_ = 10;

__device__ __forceinline__ unsigned short f2bf(float f) {
  unsigned int u = __float_as_uint(f);
  unsigned int r = u + 0x7fffu + ((u >> 16) & 1u);   // round-to-nearest-even
  return (unsigned short)(r >> 16);
}

__device__ __forceinline__ float artanh_(float x) {
  x = fminf(fmaxf(x, -1.f + 1e-7f), 1.f - 1e-7f);
  return 0.5f * (log1pf(x) - log1pf(-x));
}

__device__ __forceinline__ float sigmoid_(float x) { return 1.f / (1.f + expf(-x)); }

// Block-wide sum over 256 threads using plain LDS tree (safe, reusable).
__device__ __forceinline__ float blockSum(float v, float* s) {
  const int tid = threadIdx.x;
  __syncthreads();
  s[tid] = v;
  __syncthreads();
  for (int off = 128; off > 0; off >>= 1) {
    if (tid < off) s[tid] += s[tid + off];
    __syncthreads();
  }
  return s[0];
}

// ---------------------------------------------------------------------------
// bf16 NT GEMM:  C[M,N] = A[M,K] * B[N,K]^T  (+bias, optional ReLU, bf16 copy)
// Block = 256 thr (8 waves). Block tile 64x64; each wave: 16 rows x 32 cols
// via two v_wmma_f32_16x16x32_bf16 accumulators. Fragments loaded directly
// from global (weights are L2-resident on MI455X: 192MB L2 >> 25MB weights).
// A fragment (16-bit A 16x32): lane<16 holds row=lane, K 0..15 contiguous;
//                              lane>=16 holds row=lane-16, K 16..31.
// B fragment (16-bit B 32x16): lane<16 holds col=lane, K 0..15 contiguous;
//                              lane>=16 holds col=lane-16, K 16..31.
// ---------------------------------------------------------------------------
__global__ __launch_bounds__(256)
void gemm_bf16_nt(const unsigned short* __restrict__ A,
                  const unsigned short* __restrict__ Bw,
                  float* __restrict__ C,
                  unsigned short* __restrict__ Cbf,   // may be null
                  const float* __restrict__ bias,     // may be null
                  int M, int N, int K, int act) {
  const int tid   = threadIdx.x;
  const int lane  = tid & 31;
  const int wave  = tid >> 5;
  const int waveM = wave & 3;
  const int waveN = wave >> 2;
  const int row0  = blockIdx.y * 64 + waveM * 16;
  const int col0  = blockIdx.x * 64 + waveN * 32;
  const int r     = lane & 15;
  const int hi    = lane >> 4;       // 0 or 1
  const int akoff = hi * 16;
  const int bkoff = hi * 16;

  const unsigned short* Ap  = A  + (size_t)(row0 + r) * K;
  const unsigned short* Bp0 = Bw + (size_t)(col0 + r) * K;
  const unsigned short* Bp1 = Bw + (size_t)(col0 + 16 + r) * K;

  v8f acc0 = {0.f, 0.f, 0.f, 0.f, 0.f, 0.f, 0.f, 0.f};
  v8f acc1 = acc0;

  for (int k = 0; k < K; k += 32) {
    ushort8 a_lo  = *(const ushort8*)(Ap  + k + akoff);
    ushort8 a_hi  = *(const ushort8*)(Ap  + k + akoff + 8);
    ushort8 b0_lo = *(const ushort8*)(Bp0 + k + bkoff);
    ushort8 b0_hi = *(const ushort8*)(Bp0 + k + bkoff + 8);
    ushort8 b1_lo = *(const ushort8*)(Bp1 + k + bkoff);
    ushort8 b1_hi = *(const ushort8*)(Bp1 + k + bkoff + 8);
    BF16Frag fa, fb0, fb1;
    fa.u  = SHUF16(a_lo, a_hi);
    fb0.u = SHUF16(b0_lo, b0_hi);
    fb1.u = SHUF16(b1_lo, b1_hi);
    acc0 = __builtin_amdgcn_wmma_f32_16x16x32_bf16(false, fa.b, false, fb0.b,
                                                   (short)0, acc0, false, false);
    acc1 = __builtin_amdgcn_wmma_f32_16x16x32_bf16(false, fa.b, false, fb1.b,
                                                   (short)0, acc1, false, false);
  }

  // C/D layout: VGPR v -> M = v + 8*hi (lanes>=16 add 8), N = lane&15.
  const int mbase = row0 + hi * 8;
  const int n0 = col0 + r;
  const int n1 = n0 + 16;
  const float bi0 = bias ? bias[n0] : 0.f;
  const float bi1 = bias ? bias[n1] : 0.f;
#pragma unroll
  for (int v = 0; v < 8; ++v) {
    const int m = mbase + v;
    float x0 = acc0[v] + bi0;
    float x1 = acc1[v] + bi1;
    if (act == 1) { x0 = fmaxf(x0, 0.f); x1 = fmaxf(x1, 0.f); }
    C[(size_t)m * N + n0] = x0;
    C[(size_t)m * N + n1] = x1;
    if (Cbf) {
      Cbf[(size_t)m * N + n0] = f2bf(x0);
      Cbf[(size_t)m * N + n1] = f2bf(x1);
    }
  }
}

// ---------------------------------------------------------------------------
// Weight conversion kernels
// ---------------------------------------------------------------------------
__global__ __launch_bounds__(256)
void cvt_bf16(const float* __restrict__ in, unsigned short* __restrict__ out, int n) {
  int i = blockIdx.x * 256 + threadIdx.x;
  if (i < n) out[i] = f2bf(in[i]);
}

// pack w_hh rows [0:1024) (W_hr) and [2048:3072) (W_hz) into [2048,1024] bf16
__global__ __launch_bounds__(256)
void cvt_pack_rz(const float* __restrict__ w_hh, unsigned short* __restrict__ out) {
  int i = blockIdx.x * 256 + threadIdx.x;            // 0 .. 2048*1024-1
  if (i < 2048 * HID_) {
    int row = i >> 10;
    int src = (row < HID_) ? row : (row + HID_);
    out[i] = f2bf(w_hh[(size_t)src * HID_ + (i & 1023)]);
  }
}

// ---------------------------------------------------------------------------
// xs = expmap0(transpose(sentence_feats)) -> bf16 + per-row norm tanh(||u||)
// one block per (t,b) row
// ---------------------------------------------------------------------------
__global__ __launch_bounds__(256)
void prep_xs(const float* __restrict__ feats, unsigned short* __restrict__ xs_bf,
             float* __restrict__ xn) {
  __shared__ float s[256];
  const int row = blockIdx.x;           // row = t*B + b
  const int t = row / B_, b = row % B_;
  const int tid = threadIdx.x;
  const float* u = feats + ((size_t)b * T_ + t) * HID_;
  float v[4]; float n2 = 0.f;
#pragma unroll
  for (int i = 0; i < 4; ++i) { v[i] = u[tid + 256 * i]; n2 += v[i] * v[i]; }
  n2 = blockSum(n2, s);
  const float n  = sqrtf(fmaxf(n2, 1e-30f));
  const float sc = tanhf(n) / n;
#pragma unroll
  for (int i = 0; i < 4; ++i)
    xs_bf[(size_t)row * HID_ + tid + 256 * i] = f2bf(sc * v[i]);
  if (tid == 0) xn[row] = fmaxf(tanhf(n), 1e-15f);
}

__global__ __launch_bounds__(256)
void init_state(float* h, unsigned short* h_bf, float* hn, unsigned short* inp_bf) {
  int i = blockIdx.x * 256 + threadIdx.x;
  if (i < B_ * HID_) { h[i] = 0.f; h_bf[i] = 0; inp_bf[i] = 0; }
  if (i < B_) hn[i] = 1e-15f;
}

// ---------------------------------------------------------------------------
// one_rnn_transform (Mobius) given precomputed GEMM results; optional logmap0
// ---------------------------------------------------------------------------
__device__ __forceinline__ void mobius_gate(const float* __restrict__ Wh_p,
                                            const float* __restrict__ Ux_p,
                                            const float* __restrict__ b_p,
                                            float pn, float xnv, float* sbuf,
                                            int tid, float out[4], bool do_log) {
  float Wh[4], Ux[4], bv[4];
  float mW2 = 0.f, mU2 = 0.f;
#pragma unroll
  for (int i = 0; i < 4; ++i) {
    const int c = tid + 256 * i;
    Wh[i] = Wh_p[c]; Ux[i] = Ux_p[c]; bv[i] = b_p[c];
    mW2 += Wh[i] * Wh[i]; mU2 += Ux[i] * Ux[i];
  }
  mW2 = blockSum(mW2, sbuf);
  mU2 = blockSum(mU2, sbuf);
  const float mWn = sqrtf(fmaxf(mW2, 1e-30f));
  const float mUn = sqrtf(fmaxf(mU2, 1e-30f));
  const float sW = tanhf(mWn / pn * artanh_(pn)) / mWn;   // mobius_matvec scale
  const float sU = tanhf(mUn / xnv * artanh_(xnv)) / mUn;
  const float x2 = sW * sW * mW2, y2 = sU * sU * mU2;
  float xy = 0.f;
#pragma unroll
  for (int i = 0; i < 4; ++i) { Wh[i] *= sW; Ux[i] *= sU; xy += Wh[i] * Ux[i]; }
  xy = blockSum(xy, sbuf);
  const float den = fmaxf(1.f + 2.f * xy + x2 * y2, 1e-15f);
  const float cA = (1.f + 2.f * xy + y2) / den, cB = (1.f - x2) / den;
  float a[4]; float a2 = 0.f, b2 = 0.f, ab = 0.f;
#pragma unroll
  for (int i = 0; i < 4; ++i) {
    a[i] = cA * Wh[i] + cB * Ux[i];
    a2 += a[i] * a[i]; b2 += bv[i] * bv[i]; ab += a[i] * bv[i];
  }
  a2 = blockSum(a2, sbuf); b2 = blockSum(b2, sbuf); ab = blockSum(ab, sbuf);
  const float den2 = fmaxf(1.f + 2.f * ab + a2 * b2, 1e-15f);
  const float c1 = (1.f + 2.f * ab + b2) / den2, c2v = (1.f - a2) / den2;
  float v2 = 0.f;
#pragma unroll
  for (int i = 0; i < 4; ++i) { out[i] = c1 * a[i] + c2v * bv[i]; v2 += out[i] * out[i]; }
  if (do_log) {                       // logmap0
    v2 = blockSum(v2, sbuf);
    const float vn = sqrtf(fmaxf(v2, 1e-30f));
    const float sc = artanh_(vn) / vn;
#pragma unroll
    for (int i = 0; i < 4; ++i) out[i] *= sc;
  }
}

// gates: z, r (sigmoid(logmap0(one_rnn_transform))) and rh = mobius_pw_mul(r,h)
__global__ __launch_bounds__(256)
void enc_gates(const float* __restrict__ Gx, const float* __restrict__ Ghrz,
               const float* __restrict__ h, const float* __restrict__ hn,
               const float* __restrict__ xn, const float* __restrict__ b_hyp,
               float* __restrict__ z_out, unsigned short* __restrict__ rh_bf,
               float* __restrict__ rhn_out) {
  __shared__ float s[256];
  const int b = blockIdx.x, tid = threadIdx.x;
  const float pn = hn[b], xv = xn[b];
  float zlm[4], rlm[4];
  // z gate: W_hz (packed row 1) x h, W_iz (Gx cols 2048:3072), b_z = b_hyp[2]
  mobius_gate(Ghrz + (size_t)b * 2048 + 1024, Gx + (size_t)b * G3_ + 2048,
              b_hyp + 2 * HID_, pn, xv, s, tid, zlm, true);
  // r gate: W_hr (packed row 0) x h, W_ir (Gx cols 0:1024), b_r = b_hyp[0]
  mobius_gate(Ghrz + (size_t)b * 2048, Gx + (size_t)b * G3_,
              b_hyp, pn, xv, s, tid, rlm, true);
  float rv[4], hv[4], wx[4];
  float wx2 = 0.f;
#pragma unroll
  for (int i = 0; i < 4; ++i) {
    const int c = tid + 256 * i;
    const float zz = sigmoid_(zlm[i]);
    rv[i] = sigmoid_(rlm[i]);
    z_out[(size_t)b * HID_ + c] = zz;
    hv[i] = h[(size_t)b * HID_ + c];
    wx[i] = rv[i] * hv[i];
    wx2 += wx[i] * wx[i];
  }
  wx2 = blockSum(wx2, s);
  const float wxn = sqrtf(fmaxf(wx2, 1e-30f));
  const float sc = tanhf(wxn / pn * artanh_(pn)) / wxn;   // mobius_pw_mul scale
#pragma unroll
  for (int i = 0; i < 4; ++i)
    rh_bf[(size_t)b * HID_ + tid + 256 * i] = f2bf(sc * wx[i]);
  if (tid == 0) rhn_out[b] = fmaxf(sc * wxn, 1e-15f);
}

// combine: h_tilde, delta = mobius_add(-h,h_tilde), h' = mobius_add(h, z (x) delta)
__global__ __launch_bounds__(256)
void enc_combine(const float* __restrict__ Gx, const float* __restrict__ Ght,
                 const float* __restrict__ z, const float* __restrict__ rhn,
                 const float* __restrict__ xn, const float* __restrict__ b_hyp,
                 float* __restrict__ h, unsigned short* __restrict__ h_bf,
                 float* __restrict__ hn) {
  __shared__ float s[256];
  const int b = blockIdx.x, tid = threadIdx.x;
  float ht[4];
  mobius_gate(Ght + (size_t)b * HID_, Gx + (size_t)b * G3_ + 1024,
              b_hyp + HID_, rhn[b], xn[b], s, tid, ht, false);
  float hv[4];
  float h2 = 0.f, t2 = 0.f, hxt = 0.f;
#pragma unroll
  for (int i = 0; i < 4; ++i) {
    hv[i] = h[(size_t)b * HID_ + tid + 256 * i];
    h2 += hv[i] * hv[i]; t2 += ht[i] * ht[i]; hxt += (-hv[i]) * ht[i];
  }
  h2 = blockSum(h2, s); t2 = blockSum(t2, s); hxt = blockSum(hxt, s);
  // delta = mobius_add(-h, h_tilde)
  {
    const float den = fmaxf(1.f + 2.f * hxt + h2 * t2, 1e-15f);
    const float c1 = (1.f + 2.f * hxt + t2) / den, c2v = (1.f - h2) / den;
    float dl[4]; float d2 = 0.f;
#pragma unroll
    for (int i = 0; i < 4; ++i) { dl[i] = c1 * (-hv[i]) + c2v * ht[i]; d2 += dl[i] * dl[i]; }
    d2 = blockSum(d2, s);
    const float dn = sqrtf(fmaxf(d2, 1e-30f));
    // zd = mobius_pw_mul(z, delta)
    float wx[4]; float wx2 = 0.f;
#pragma unroll
    for (int i = 0; i < 4; ++i) {
      wx[i] = z[(size_t)b * HID_ + tid + 256 * i] * dl[i];
      wx2 += wx[i] * wx[i];
    }
    wx2 = blockSum(wx2, s);
    const float wxn = sqrtf(fmaxf(wx2, 1e-30f));
    const float sc = tanhf(wxn / dn * artanh_(dn)) / wxn;
    float zd[4]; float zd2 = 0.f, hzd = 0.f;
#pragma unroll
    for (int i = 0; i < 4; ++i) { zd[i] = sc * wx[i]; zd2 += zd[i] * zd[i]; hzd += hv[i] * zd[i]; }
    zd2 = blockSum(zd2, s); hzd = blockSum(hzd, s);
    // h_new = mobius_add(h, zd)
    const float kd = fmaxf(1.f + 2.f * hzd + h2 * zd2, 1e-15f);
    const float k1 = (1.f + 2.f * hzd + zd2) / kd, k2 = (1.f - h2) / kd;
    float hn2 = 0.f;
#pragma unroll
    for (int i = 0; i < 4; ++i) {
      const float v = k1 * hv[i] + k2 * zd[i];
      const int c = tid + 256 * i;
      h[(size_t)b * HID_ + c] = v;
      h_bf[(size_t)b * HID_ + c] = f2bf(v);
      hn2 += v * v;
    }
    hn2 = blockSum(hn2, s);
    if (tid == 0) hn[b] = sqrtf(fmaxf(hn2, 1e-30f));
  }
}

// h := logmap0(h) (in place), refresh bf16 copy
__global__ __launch_bounds__(256)
void logmap_h(float* __restrict__ h, unsigned short* __restrict__ h_bf) {
  __shared__ float s[256];
  const int b = blockIdx.x, tid = threadIdx.x;
  float v[4]; float n2 = 0.f;
#pragma unroll
  for (int i = 0; i < 4; ++i) { v[i] = h[(size_t)b * HID_ + tid + 256 * i]; n2 += v[i] * v[i]; }
  n2 = blockSum(n2, s);
  const float n = sqrtf(fmaxf(n2, 1e-30f));
  const float sc = artanh_(n) / n;
#pragma unroll
  for (int i = 0; i < 4; ++i) {
    const float o = sc * v[i];
    const int c = tid + 256 * i;
    h[(size_t)b * HID_ + c] = o;
    h_bf[(size_t)b * HID_ + c] = f2bf(o);
  }
}

// softmax head: out[b*rowStride + j] = softmax_j(h[b] . W[j] + bias[j])
__global__ __launch_bounds__(256)
void head_softmax(const float* __restrict__ h, const float* __restrict__ W,
                  const float* __restrict__ bias, float* __restrict__ out,
                  int ncols, int rowStride) {
  __shared__ float s[256];
  __shared__ float dots[8];
  const int b = blockIdx.x, tid = threadIdx.x;
  float hv[4];
#pragma unroll
  for (int i = 0; i < 4; ++i) hv[i] = h[(size_t)b * HID_ + tid + 256 * i];
  for (int j = 0; j < ncols; ++j) {
    float p = 0.f;
#pragma unroll
    for (int i = 0; i < 4; ++i) p += hv[i] * W[(size_t)j * HID_ + tid + 256 * i];
    p = blockSum(p, s);
    if (tid == 0) dots[j] = p + bias[j];
  }
  __syncthreads();
  if (tid == 0) {
    float mx = -1e30f;
    for (int j = 0; j < ncols; ++j) mx = fmaxf(mx, dots[j]);
    float se = 0.f; float e[8];
    for (int j = 0; j < ncols; ++j) { e[j] = expf(dots[j] - mx); se += e[j]; }
    for (int j = 0; j < ncols; ++j) out[(size_t)b * rowStride + j] = e[j] / se;
  }
}

// torch GRU cell elementwise (biases already added in GEMM epilogue)
__global__ __launch_bounds__(256)
void dec_gru(const float* __restrict__ gi, const float* __restrict__ gh,
             float* __restrict__ h, unsigned short* __restrict__ h_bf) {
  const int i = blockIdx.x * 256 + threadIdx.x;   // 0 .. B*HID-1
  const int b = i >> 10, k = i & 1023;
  const float* gib = gi + (size_t)b * G3_;
  const float* ghb = gh + (size_t)b * G3_;
  const float r = sigmoid_(gib[k] + ghb[k]);
  const float z = sigmoid_(gib[1024 + k] + ghb[1024 + k]);
  const float n = tanhf(gib[2048 + k] + r * ghb[2048 + k]);
  const float hnew = (1.f - z) * n + z * h[i];
  h[i] = hnew;
  h_bf[i] = f2bf(hnew);
}

// ---------------------------------------------------------------------------
// Host-side orchestration (all launches on `stream`; graph-capture safe)
// ---------------------------------------------------------------------------
extern "C" void kernel_launch(void* const* d_in, const int* in_sizes, int n_in,
                              void* d_out, int out_size, void* d_ws, size_t ws_size,
                              hipStream_t stream) {
  (void)in_sizes; (void)n_in; (void)out_size; (void)ws_size;
  const float* feats    = (const float*)d_in[0];
  const float* w_ih     = (const float*)d_in[1];
  const float* w_hh     = (const float*)d_in[2];
  const float* b_hyp    = (const float*)d_in[3];
  const float* dec_w_ih = (const float*)d_in[4];
  const float* dec_w_hh = (const float*)d_in[5];
  const float* dec_b_ih = (const float*)d_in[6];
  const float* dec_b_hh = (const float*)d_in[7];
  const float* fc_in_w  = (const float*)d_in[8];
  const float* fc_in_b  = (const float*)d_in[9];
  const float* fc_out_w = (const float*)d_in[10];
  const float* fc_out_b = (const float*)d_in[11];
  const float* span_w   = (const float*)d_in[12];
  const float* span_b   = (const float*)d_in[13];
  float* out = (float*)d_out;

  // bump allocator over workspace (256B aligned regions)
  char* p = (char*)d_ws;
  auto alloc = [&](size_t bytes) -> void* {
    void* r = (void*)p;
    p += (bytes + 255) & ~(size_t)255;
    return r;
  };
  unsigned short* wih_bf   = (unsigned short*)alloc((size_t)G3_ * HID_ * 2);
  unsigned short* whh_bf   = (unsigned short*)alloc((size_t)G3_ * HID_ * 2);
  unsigned short* whh_rz   = (unsigned short*)alloc((size_t)2048 * HID_ * 2);
  unsigned short* dwih_bf  = (unsigned short*)alloc((size_t)G3_ * HID_ * 2);
  unsigned short* dwhh_bf  = (unsigned short*)alloc((size_t)G3_ * HID_ * 2);
  unsigned short* fcin_bf  = (unsigned short*)alloc((size_t)HID_ * HID_ * 2);
  unsigned short* xs_bf    = (unsigned short*)alloc((size_t)T_ * B_ * HID_ * 2);
  float*          xn       = (float*)alloc((size_t)T_ * B_ * 4);
  float*          Gx       = (float*)alloc((size_t)B_ * G3_ * 4);
  float*          Ghrz     = (float*)alloc((size_t)B_ * 2048 * 4);
  float*          Ght      = (float*)alloc((size_t)B_ * HID_ * 4);
  float*          h        = (float*)alloc((size_t)B_ * HID_ * 4);
  unsigned short* h_bf     = (unsigned short*)alloc((size_t)B_ * HID_ * 2);
  float*          hn       = (float*)alloc((size_t)B_ * 4);
  float*          z        = (float*)alloc((size_t)B_ * HID_ * 4);
  unsigned short* rh_bf    = (unsigned short*)alloc((size_t)B_ * HID_ * 2);
  float*          rhn      = (float*)alloc((size_t)B_ * 4);
  float*          gi       = (float*)alloc((size_t)B_ * G3_ * 4);
  float*          gh       = (float*)alloc((size_t)B_ * G3_ * 4);
  float*          inp      = (float*)alloc((size_t)B_ * HID_ * 4);
  unsigned short* inp_bf   = (unsigned short*)alloc((size_t)B_ * HID_ * 2);

  const int NWH = G3_ * HID_;
  cvt_bf16<<<(NWH + 255) / 256, 256, 0, stream>>>(w_ih, wih_bf, NWH);
  cvt_bf16<<<(NWH + 255) / 256, 256, 0, stream>>>(w_hh, whh_bf, NWH);
  cvt_pack_rz<<<(2048 * HID_ + 255) / 256, 256, 0, stream>>>(w_hh, whh_rz);
  cvt_bf16<<<(NWH + 255) / 256, 256, 0, stream>>>(dec_w_ih, dwih_bf, NWH);
  cvt_bf16<<<(NWH + 255) / 256, 256, 0, stream>>>(dec_w_hh, dwhh_bf, NWH);
  cvt_bf16<<<(HID_ * HID_ + 255) / 256, 256, 0, stream>>>(fc_in_w, fcin_bf, HID_ * HID_);

  prep_xs<<<T_ * B_, 256, 0, stream>>>(feats, xs_bf, xn);
  init_state<<<(B_ * HID_ + 255) / 256, 256, 0, stream>>>(h, h_bf, hn, inp_bf);

  // ---- encoder scan (75 dependent steps) ----
  for (int t = 0; t < T_; ++t) {
    const unsigned short* x_bf = xs_bf + (size_t)t * B_ * HID_;
    const float* xn_t = xn + (size_t)t * B_;
    // Gx = x @ w_ih^T   [256,3072]
    gemm_bf16_nt<<<dim3(G3_ / 64, B_ / 64), 256, 0, stream>>>(
        x_bf, wih_bf, Gx, nullptr, nullptr, B_, G3_, HID_, 0);
    // Ghrz = h @ [W_hr;W_hz]^T  [256,2048]
    gemm_bf16_nt<<<dim3(2048 / 64, B_ / 64), 256, 0, stream>>>(
        h_bf, whh_rz, Ghrz, nullptr, nullptr, B_, 2048, HID_, 0);
    enc_gates<<<B_, 256, 0, stream>>>(Gx, Ghrz, h, hn, xn_t, b_hyp, z, rh_bf, rhn);
    // Ght = rh @ W_hh_^T  [256,1024]  (w_hh rows 1024:2048)
    gemm_bf16_nt<<<dim3(HID_ / 64, B_ / 64), 256, 0, stream>>>(
        rh_bf, whh_bf + (size_t)HID_ * HID_, Ght, nullptr, nullptr, B_, HID_, HID_, 0);
    enc_combine<<<B_, 256, 0, stream>>>(Gx, Ght, z, rhn, xn_t, b_hyp, h, h_bf, hn);
  }

  // h_out = logmap0(h_last); span head
  logmap_h<<<B_, 256, 0, stream>>>(h, h_bf);
  head_softmax<<<B_, 256, 0, stream>>>(h, span_w, span_b, out, 4, 4);

  // ---- decoder (10 steps) ----
  for (int d = 0; d < NUM_DAYS_; ++d) {
    gemm_bf16_nt<<<dim3(G3_ / 64, B_ / 64), 256, 0, stream>>>(
        inp_bf, dwih_bf, gi, nullptr, dec_b_ih, B_, G3_, HID_, 0);
    gemm_bf16_nt<<<dim3(G3_ / 64, B_ / 64), 256, 0, stream>>>(
        h_bf, dwhh_bf, gh, nullptr, dec_b_hh, B_, G3_, HID_, 0);
    dec_gru<<<(B_ * HID_) / 256, 256, 0, stream>>>(gi, gh, h, h_bf);
    // outputs[b, d, :] -> d_out[1024 + (b*10 + d)*2 + j]
    head_softmax<<<B_, 256, 0, stream>>>(h, fc_out_w, fc_out_b,
                                         out + 1024 + (size_t)d * 2, 2, 2 * NUM_DAYS_);
    // inp = relu(h @ fc_in_w^T + b), bf16 copy for next GEMM
    gemm_bf16_nt<<<dim3(HID_ / 64, B_ / 64), 256, 0, stream>>>(
        h_bf, fcin_bf, inp, inp_bf, fc_in_b, B_, HID_, HID_, 1);
  }
}
